// KappaGCNLayer_69758858822202
// MI455X (gfx1250) — matrix operand order repaired and checked by reading, not compiled, not run on hardware
//
#include <hip/hip_runtime.h>
#include <hip/hip_bf16.h>
#include <math.h>

// ---------------------------------------------------------------------------
// Hyperbolic (Poincare ball) GCN layer for MI455X / gfx1250.
//
// Roofline: dominant cost is A_hat (8192x8192 fp32 = 256MB) read once from
// HBM -> ~11.5us floor at 23.3 TB/s. 17.4 GFLOP of GEMM => 65 FLOP/B, so we
// use bf16 WMMA (16x16x32) for the aggregation GEMM to stay bandwidth-bound,
// and f32 WMMA (16x16x4) for the small, precision-sensitive X@W.
// denominator (A@(gamma-1)) and row-sums (A@1) are fused as extra B columns.
// B panels are double-buffered through LDS with gfx1250 async global->LDS
// copies (ASYNCcnt), overlapping the next panel's fetch with WMMA compute.
// ---------------------------------------------------------------------------

typedef __attribute__((ext_vector_type(2)))  float  v2f;
typedef __attribute__((ext_vector_type(8)))  float  v8f;
typedef __attribute__((ext_vector_type(16))) __bf16 v16bf;
typedef __attribute__((ext_vector_type(4)))  int    v4i;

// Pointer types matching the async-LDS builtin's parameters:
//   param0: v4i in global (AS1, printed '__device__') ; param1: v4i in LDS (AS3)
typedef v4i __attribute__((address_space(1)))* gv4i_ptr;
typedef v4i __attribute__((address_space(3)))* lv4i_ptr;

#define N_NODES 8192
#define DIM     128
#define NT      9            // 144 cols / 16  (128 data + denom + rowsum + pad)
#define KTILES  (N_NODES/32) // 256 K-steps of 32 for the big GEMM
#define BP_KT_USHORTS (NT * 32 * 16)          // 4608 ushorts = 9216 B per K-step
#define BP_KT_CHUNKS  (NT * 32 * 2)           // 576 16-byte chunks per K-step

#if defined(__has_builtin)
# if __has_builtin(__builtin_amdgcn_global_load_async_to_lds_b128) && \
     __has_builtin(__builtin_amdgcn_s_wait_asynccnt)
#  define HAVE_ASYNC_LDS 1
# endif
#endif
#ifndef HAVE_ASYNC_LDS
# define HAVE_ASYNC_LDS 0
#endif

__device__ __forceinline__ float artanh_clip(float x) {
    x = fminf(fmaxf(x, -1.0f + 1e-5f), 1.0f - 1e-5f);
    return 0.5f * logf((1.0f + x) / (1.0f - x));
}

__device__ __forceinline__ float red16(float v) {   // reduce within 16-lane half
    #pragma unroll
    for (int m = 1; m < 16; m <<= 1) v += __shfl_xor(v, m, 32);
    return v;
}
__device__ __forceinline__ float red32(float v) {   // full wave32 reduce
    #pragma unroll
    for (int m = 1; m < 32; m <<= 1) v += __shfl_xor(v, m, 32);
    return v;
}

__device__ __forceinline__ unsigned short f2bf(float f) {  // RNE float->bf16 bits
    union { float f; unsigned int u; } v; v.f = f;
    unsigned int r = v.u + 0x7FFFu + ((v.u >> 16) & 1u);
    return (unsigned short)(r >> 16);
}

// ---------------------------------------------------------------------------
// Kernel 1: MX = X @ W   (8192x128x128, fp32 WMMA 16x16x4)
// ---------------------------------------------------------------------------
__global__ __launch_bounds__(256)
void k_xw(const float* __restrict__ X, const float* __restrict__ W,
          float* __restrict__ MX) {
    const int wave = threadIdx.x >> 5;       // N-tile
    const int lane = threadIdx.x & 31;
    const int h = lane >> 4, l = lane & 15;
    const int m0 = blockIdx.x * 16;
    const int n0 = wave * 16;

    v8f c = {};
    #pragma unroll 4
    for (int k = 0; k < DIM; k += 4) {
        // A operand 16x4 f32: lane holds M=l, K = {2h, 2h+1}
        const float2 a2 = *reinterpret_cast<const float2*>(X + (size_t)(m0 + l) * DIM + k + 2 * h);
        v2f a; a[0] = a2.x; a[1] = a2.y;
        // B operand 4x16 f32: lane holds N=l, K = {2h, 2h+1}
        v2f b;
        b[0] = W[(size_t)(k + 2 * h) * DIM + n0 + l];
        b[1] = W[(size_t)(k + 2 * h + 1) * DIM + n0 + l];
        c = __builtin_amdgcn_wmma_f32_16x16x4_f32(false, a, false, b,
                                                  (short)0, c, false, false);
    }
    #pragma unroll
    for (int i = 0; i < 8; i++)
        MX[(size_t)(m0 + h * 8 + i) * DIM + n0 + l] = c[i];
}

// ---------------------------------------------------------------------------
// Kernel 2: mobius_matvec rescale + gamma; emit B in bf16 *pre-swizzled into
// the WMMA B-operand register layout*:
//   Bpack[(kt*NT + nt)*32 + lane][q], lane = 16*((j%32)/16) + (col%16), q = j%16
// cols: 0..127 = gamma*XW, 128 = gamma-1, 129 = 1 (row-sum column), 130..143 = 0
// ---------------------------------------------------------------------------
__global__ __launch_bounds__(256)
void k_pack(const float* __restrict__ X, const float* __restrict__ MX,
            unsigned short* __restrict__ Bp) {
    const int row  = blockIdx.x * 8 + (threadIdx.x >> 5);
    const int lane = threadIdx.x & 31;

    float x[4], mx[4], xs = 0.f, ms = 0.f;
    #pragma unroll
    for (int i = 0; i < 4; i++) {
        const int c = lane + 32 * i;
        x[i]  = X [(size_t)row * DIM + c];
        mx[i] = MX[(size_t)row * DIM + c];
        xs += x[i] * x[i];
        ms += mx[i] * mx[i];
    }
    xs = red32(xs); ms = red32(ms);
    const float xn  = fmaxf(sqrtf(xs), 1e-15f);
    const float mxn = fmaxf(sqrtf(ms), 1e-15f);
    const float sc  = tanhf(mxn / xn * artanh_clip(xn)) / mxn;   // res = sc * mx

    float res[4], sq = 0.f;
    #pragma unroll
    for (int i = 0; i < 4; i++) { res[i] = sc * mx[i]; sq += res[i] * res[i]; }
    sq = red32(sq);
    const float gamma = 2.0f / fmaxf(1.0f - sq, 1e-5f);

    const size_t kt = row >> 5;
    const int h2 = (row >> 4) & 1;
    const int q  = row & 15;
    #pragma unroll
    for (int i = 0; i < 4; i++) {
        const int c  = lane + 32 * i;
        const int nt = c >> 4;
        const int ls = h2 * 16 + (c & 15);
        Bp[(kt * BP_KT_USHORTS) + (nt * 32 + ls) * 16 + q] = f2bf(gamma * res[i]);
    }
    if (lane < 16) {                       // cols 128..143: denom, rowsum, pad
        const int c  = 128 + lane;
        const float v = (lane == 0) ? (gamma - 1.0f) : (lane == 1 ? 1.0f : 0.0f);
        const int nt = c >> 4;             // == 8
        const int ls = h2 * 16 + (c & 15);
        Bp[(kt * BP_KT_USHORTS) + (nt * 32 + ls) * 16 + q] = f2bf(v);
    }
}

// ---------------------------------------------------------------------------
// Kernel 3: C = A_hat @ B (8192 x 144 x K=8192) with bf16 WMMA, fused
// hyperbolic epilogue. 64 blocks x 8 waves; wave owns 16 rows, 9 accumulators.
// B panel (9216 B / K-step) double-buffered in LDS via async global->LDS.
// ---------------------------------------------------------------------------
#if HAVE_ASYNC_LDS
__device__ __forceinline__ void stage_B_async(const unsigned short* Bp, int kt,
                                              unsigned short* dstLds, int tid) {
    const char* src = (const char*)(Bp + (size_t)kt * BP_KT_USHORTS);
    char* dst = (char*)dstLds;
    #pragma unroll
    for (int it = 0; it < 3; ++it) {
        const int idx = tid + it * 256;
        if (idx < BP_KT_CHUNKS) {
            __builtin_amdgcn_global_load_async_to_lds_b128(
                (gv4i_ptr)(unsigned long long)(src + idx * 16),
                (lv4i_ptr)(unsigned int)(unsigned long long)(dst + idx * 16),
                0, 0);
        }
    }
}
#endif

__global__ __launch_bounds__(256)
void k_agg(const float* __restrict__ A, const unsigned short* __restrict__ Bp,
           float* __restrict__ out) {
    const int tid  = threadIdx.x;
    const int wave = tid >> 5;
    const int lane = tid & 31;
    const int h = lane >> 4, l = lane & 15;
    const int mbase = blockIdx.x * 128 + wave * 16;
    const float* Arow = A + (size_t)(mbase + l) * N_NODES;

#if HAVE_ASYNC_LDS
    __shared__ unsigned short Bs[2][BP_KT_USHORTS];   // 2 x 9216 B
    stage_B_async(Bp, 0, Bs[0], tid);
#endif

    v8f acc[NT] = {};

    for (int kt = 0; kt < KTILES; ++kt) {
#if HAVE_ASYNC_LDS
        const int cur = kt & 1;
        __builtin_amdgcn_s_wait_asynccnt(0);   // own panel writes landed
        __syncthreads();                       // everyone's landed; prev readers done
        if (kt + 1 < KTILES)
            stage_B_async(Bp, kt + 1, Bs[cur ^ 1], tid);   // overlaps compute below
#endif
        // A operand 16x32 bf16: lane (M=l, half h) holds K = {16h..16h+15}
        const int k0 = kt * 32 + 16 * h;
        const float4 fa0 = *reinterpret_cast<const float4*>(Arow + k0);
        const float4 fa1 = *reinterpret_cast<const float4*>(Arow + k0 + 4);
        const float4 fb0 = *reinterpret_cast<const float4*>(Arow + k0 + 8);
        const float4 fb1 = *reinterpret_cast<const float4*>(Arow + k0 + 12);
        v16bf a;
        a[0]  = (__bf16)fa0.x; a[1]  = (__bf16)fa0.y; a[2]  = (__bf16)fa0.z; a[3]  = (__bf16)fa0.w;
        a[4]  = (__bf16)fa1.x; a[5]  = (__bf16)fa1.y; a[6]  = (__bf16)fa1.z; a[7]  = (__bf16)fa1.w;
        a[8]  = (__bf16)fb0.x; a[9]  = (__bf16)fb0.y; a[10] = (__bf16)fb0.z; a[11] = (__bf16)fb0.w;
        a[12] = (__bf16)fb1.x; a[13] = (__bf16)fb1.y; a[14] = (__bf16)fb1.z; a[15] = (__bf16)fb1.w;

        if (kt + 8 < KTILES)
            __builtin_prefetch(Arow + (size_t)(kt + 8) * 32 + 8 * h, 0, 1);

        #pragma unroll
        for (int n = 0; n < NT; n++) {
            union { uint4 u[2]; v16bf v; } bb;
#if HAVE_ASYNC_LDS
            const uint4* bsrc = reinterpret_cast<const uint4*>(
                &Bs[cur][(n * 32 + lane) * 16]);             // ds_load_b128 x2
            bb.u[0] = bsrc[0]; bb.u[1] = bsrc[1];
#else
            const uint4* bsrc = reinterpret_cast<const uint4*>(
                Bp + (size_t)kt * BP_KT_USHORTS + (n * 32 + lane) * 16);
            bb.u[0] = bsrc[0]; bb.u[1] = bsrc[1];
#endif
            acc[n] = __builtin_amdgcn_wmma_f32_16x16x32_bf16(
                false, a, false, bb.v, (short)0, acc[n], false, false);
        }
    }

    // Fused epilogue. C layout: acc[n][j] = C[row = mbase + 8h + j][col = 16n + l]
    // acc[8][.]: col128 (lane%16==0) = denominator, col129 (==1) = row-sum r.
    #pragma unroll
    for (int j = 0; j < 8; j++) {
        float den = __shfl(acc[8][j], (lane & 16),     32);
        float r   = __shfl(acc[8][j], (lane & 16) | 1, 32);
        den = (den >= 0.0f) ? fmaxf(den, 1e-10f) : fminf(den, -1e-10f); // _clamp_abs

        float t[8], tp = 0.f;
        #pragma unroll
        for (int n = 0; n < 8; n++) { t[n] = acc[n][j] / den; tp += t[n] * t[n]; }
        tp = red16(tp);
        const float tn = fmaxf(sqrtf(tp), 1e-15f);

        const float s1 = tanhf(0.5f * artanh_clip(tn));      // ||m||, m = s1/tn * t
        const float mn = fmaxf(s1, 1e-15f);
        const float s2 = tanhf(r * artanh_clip(mn));         // ||AXW|| (signed)
        const float an = fmaxf(fabsf(s2), 1e-15f);
        const float cu = (artanh_clip(an) / an) * (s2 / mn) * (s1 / tn); // u = cu*t

        float w[8], wp = 0.f;
        #pragma unroll
        for (int n = 0; n < 8; n++) { w[n] = fmaxf(cu * t[n], 0.0f); wp += w[n] * w[n]; }
        wp = red16(wp);
        const float wn = fmaxf(sqrtf(wp), 1e-15f);
        const float so = tanhf(wn) / wn;                     // expmap0 scale

        const int m = mbase + h * 8 + j;
        #pragma unroll
        for (int n = 0; n < 8; n++)
            out[(size_t)m * DIM + n * 16 + l] = so * w[n];
    }
}

// ---------------------------------------------------------------------------
extern "C" void kernel_launch(void* const* d_in, const int* in_sizes, int n_in,
                              void* d_out, int out_size, void* d_ws, size_t ws_size,
                              hipStream_t stream) {
    const float* X = (const float*)d_in[0];   // (8192,128)
    const float* A = (const float*)d_in[1];   // (8192,8192)
    const float* W = (const float*)d_in[2];   // (128,128)
    float* out = (float*)d_out;               // (8192,128)

    char* ws = (char*)d_ws;
    float* MX = (float*)ws;                                            // 4 MB
    unsigned short* Bp =
        (unsigned short*)(ws + (size_t)N_NODES * DIM * sizeof(float)); // 2.36 MB

    hipLaunchKernelGGL(k_xw,   dim3(N_NODES / 16),  dim3(256), 0, stream, X, W, MX);
    hipLaunchKernelGGL(k_pack, dim3(N_NODES / 8),   dim3(256), 0, stream, X, MX, Bp);
    hipLaunchKernelGGL(k_agg,  dim3(N_NODES / 128), dim3(256), 0, stream, A, Bp, out);
}
